// Qwen3Attention_11639361372624
// MI455X (gfx1250) — matrix operand order repaired and checked by reading, not compiled, not run on hardware
//
#include <hip/hip_runtime.h>
#include <hip/hip_bf16.h>

typedef __attribute__((ext_vector_type(16))) _Float16 v16h;
typedef __attribute__((ext_vector_type(8)))  _Float16 v8h;
typedef __attribute__((ext_vector_type(4)))  _Float16 v4h;
typedef __attribute__((ext_vector_type(2)))  __fp16   fp16x2;
typedef __attribute__((ext_vector_type(8)))  float    v8f;
typedef __attribute__((ext_vector_type(4)))  float    f32x4;
typedef __attribute__((ext_vector_type(4)))  unsigned int u32x4;
typedef __attribute__((ext_vector_type(8)))  int      i32x8;
typedef __attribute__((ext_vector_type(4)))  int      i32x4;

// Problem constants (match the reference)
constexpr int Bc   = 2;
constexpr int Qc   = 1024;
constexpr int Pc   = 3072;
constexpr int HIDc = 4096;
constexpr int Hc   = 32;
constexpr int KVc  = 8;
constexpr int Dc   = 128;
constexpr int Gc   = Hc / KVc;     // 4
constexpr int Sc   = Pc + Qc;      // 4096
constexpr float EPSc = 1e-6f;
constexpr float SCALEc = 0.08838834764831845f; // 1/sqrt(128)

// ---------------------------------------------------------------------------
// WMMA fragment helpers (wave32 layouts per CDNA5 ISA 7.12.2), vectorized:
// every lane's 16 halves are two contiguous 16-byte runs -> 2x b128 loads.
// ---------------------------------------------------------------------------

// A-matrix 16x32 f16, row-major with stride lda (lda multiple of 8).
__device__ __forceinline__ v16h load_a_frag16(const _Float16* A, int lda) {
  int lane = threadIdx.x & 31;
  int m  = lane & 15;
  int kb = (lane >> 4) * 8;
  v8h lo = *(const v8h*)(A + m * lda + kb);
  v8h hi = *(const v8h*)(A + m * lda + 16 + kb);
  v16h a;
#pragma unroll
  for (int i = 0; i < 8; ++i) { a[i] = lo[i]; a[8 + i] = hi[i]; }
  return a;
}

// B-matrix 32x16 f16 stored K-CONTIGUOUS: element [n][k] at Bp[n*ld + k].
// (Covers GEMM Bt tiles, K^T score tiles, and d-major V tiles.)
__device__ __forceinline__ v16h load_bT_frag(const _Float16* Bp, int ld) {
  int lane = threadIdx.x & 31;
  int n  = lane & 15;
  int kb = (lane >> 4) * 8;
  v8h lo = *(const v8h*)(Bp + n * ld + kb);
  v8h hi = *(const v8h*)(Bp + n * ld + 16 + kb);
  v16h b;
#pragma unroll
  for (int i = 0; i < 8; ++i) { b[i] = lo[i]; b[8 + i] = hi[i]; }
  return b;
}

__device__ __forceinline__ v8f wmma_f16(v16h a, v16h b, v8f c) {
  return __builtin_amdgcn_wmma_f32_16x16x32_f16(false, a, false, b, (short)0, c,
                                                false, false);
}

__device__ __forceinline__ v4h pk4(float a, float b, float c, float d) {
  fp16x2 lo = __builtin_amdgcn_cvt_pkrtz(a, b); // v_cvt_pk_rtz_f16_f32
  fp16x2 hi = __builtin_amdgcn_cvt_pkrtz(c, d);
  v4h r;
  r[0] = (_Float16)lo[0]; r[1] = (_Float16)lo[1];
  r[2] = (_Float16)hi[0]; r[3] = (_Float16)hi[1];
  return r;
}

// ---------------------------------------------------------------------------
// Tensor Data Mover: 2D f16 tile (tile_d0 contiguous elems x tile_d1 rows,
// row stride stride0 elems) from global -> LDS, packed contiguously.
// D# layout per CDNA5 ISA ch.8 (group0: count/lds/global/type, group1: dims).
// ---------------------------------------------------------------------------
__device__ __forceinline__ void tdm_load_2d_f16(unsigned lds_off,
                                                unsigned long long gaddr,
                                                unsigned tensor_d0,
                                                unsigned tensor_d1,
                                                unsigned tile_d0,
                                                unsigned tile_d1,
                                                unsigned long long stride0) {
  unsigned lds_u = (unsigned)__builtin_amdgcn_readfirstlane((int)lds_off);
  unsigned ga_lo = (unsigned)__builtin_amdgcn_readfirstlane((int)(gaddr & 0xFFFFFFFFull));
  unsigned ga_hi = (unsigned)__builtin_amdgcn_readfirstlane((int)(gaddr >> 32));

  u32x4 g0;
  g0[0] = 1u;                                   // count=1, user descriptor
  g0[1] = lds_u;                                // lds_addr
  g0[2] = ga_lo;                                // global_addr[31:0]
  g0[3] = (ga_hi & 0x1FFFFFFu) | (2u << 30);    // global_addr[56:32] | type=2

  i32x8 g1;
  g1[0] = (int)(1u << 16);                                          // data_size=2B
  g1[1] = (int)((tensor_d0 & 0xFFFFu) << 16);                       // tdim0[15:0]
  g1[2] = (int)((tensor_d0 >> 16) | ((tensor_d1 & 0xFFFFu) << 16)); // tdim0 hi|tdim1 lo
  g1[3] = (int)((tensor_d1 >> 16) | (tile_d0 << 16));               // tdim1 hi|tile0
  g1[4] = (int)tile_d1;                                             // tile1 (tile2=0)
  g1[5] = (int)(stride0 & 0xFFFFFFFFull);                           // stride0[31:0]
  g1[6] = (int)((stride0 >> 32) & 0xFFFFull);                       // stride0[47:32]
  g1[7] = 0;
  i32x4 z4 = {0, 0, 0, 0};
  i32x8 z8 = {0, 0, 0, 0, 0, 0, 0, 0};
  __builtin_amdgcn_tensor_load_to_lds(g0, g1, z4, z4, z8, 0);
}

__device__ __forceinline__ unsigned lds_offset_of(const void* p) {
  // flat shared address: LDS byte offset lives in the low 32 bits (ISA 10.2)
  return (unsigned)(unsigned long long)(uintptr_t)p;
}

// ---------------------------------------------------------------------------
// Kernel 1: GEMM  Out[M,N] = X[M,K] * W[K,N]   (f32 in, f32 out, f16 WMMA core)
// 256 threads (8 waves), tile 64(M) x 128(N), K-step 32.
// A staged row-major (k contiguous), B staged TRANSPOSED (k contiguous) so all
// fragment reads are b128 LDS loads.
// ---------------------------------------------------------------------------
__global__ void __launch_bounds__(256)
gemm_f16_wmma(const float* __restrict__ X, const float* __restrict__ W,
              float* __restrict__ Out, int M, int K, int N) {
  constexpr int TM = 64, TN = 128, TK = 32;
  __shared__ _Float16 As[TM][TK];    // 4 KB, [m][k]
  __shared__ _Float16 Bt[TN][TK];    // 8 KB, [n][k]

  int tile_m = blockIdx.y * TM;
  int tile_n = blockIdx.x * TN;
  int wave = threadIdx.x >> 5;
  int wm = wave >> 1, wn = wave & 1;

  v8f acc[4] = {};

  for (int k0 = 0; k0 < K; k0 += TK) {
    // A: 512 float4 tasks / 256 threads; each: b128 load + 2x cvt_pk + b64 store
#pragma unroll
    for (int j = 0; j < 2; ++j) {
      int e = threadIdx.x + j * 256;
      int r = e >> 3, c4 = (e & 7) << 2;
      f32x4 f = *(const f32x4*)&X[(size_t)(tile_m + r) * K + k0 + c4];
      *(v4h*)&As[r][c4] = pk4(f[0], f[1], f[2], f[3]);
    }
    // B: 1024 tasks of 4 k-consecutive elems; global reads coalesced across n
#pragma unroll
    for (int j = 0; j < 4; ++j) {
      int e = threadIdx.x + j * 256;
      int n = e & 127, k4 = (e >> 7) << 2;
      float f0 = W[(size_t)(k0 + k4 + 0) * N + tile_n + n];
      float f1 = W[(size_t)(k0 + k4 + 1) * N + tile_n + n];
      float f2 = W[(size_t)(k0 + k4 + 2) * N + tile_n + n];
      float f3 = W[(size_t)(k0 + k4 + 3) * N + tile_n + n];
      *(v4h*)&Bt[n][k4] = pk4(f0, f1, f2, f3);
    }
    if (k0 + TK < K)  // global_prefetch_b8 hint for the next weight tile
      __builtin_prefetch(&W[(size_t)(k0 + TK) * N + tile_n], 0, 1);
    __syncthreads();

    v16h a = load_a_frag16(&As[wm * 16][0], TK);
#pragma unroll
    for (int t = 0; t < 4; ++t) {
      v16h b = load_bT_frag(&Bt[wn * 64 + t * 16][0], TK);
      acc[t] = wmma_f16(a, b, acc[t]);
    }
    __syncthreads();
  }

  int lane = threadIdx.x & 31;
  int nn = lane & 15;
  int mh = (lane >> 4) * 8;
#pragma unroll
  for (int t = 0; t < 4; ++t)
#pragma unroll
    for (int r = 0; r < 8; ++r) {
      int m = tile_m + wm * 16 + mh + r;
      int n = tile_n + wn * 64 + t * 16 + nn;
      Out[(size_t)m * N + n] = acc[t][r];
    }
}

// ---------------------------------------------------------------------------
// Kernel 2: fused RMSNorm + RoPE.  One wave per (b,q,head) row of D=128.
// ---------------------------------------------------------------------------
__global__ void __launch_bounds__(256)
norm_rope_kernel(const float* __restrict__ xraw, const float* __restrict__ cosb,
                 const float* __restrict__ sinb, const float* __restrict__ w,
                 _Float16* __restrict__ out, int heads, int outS, int outBase,
                 float outScale, int nrows) {
  int row = blockIdx.x * 8 + (threadIdx.x >> 5);
  if (row >= nrows) return;
  int lane = threadIdx.x & 31;
  int h  = row % heads;
  int bq = row / heads;
  int q  = bq % Qc;
  int b  = bq / Qc;

  const float* x = xraw + (size_t)row * Dc;
  float v[4];
  float ss = 0.f;
#pragma unroll
  for (int i = 0; i < 4; ++i) {
    v[i] = x[lane + 32 * i];
    ss += v[i] * v[i];
  }
#pragma unroll
  for (int off = 16; off >= 1; off >>= 1) ss += __shfl_xor(ss, off, 32);
  float rinv = rsqrtf(ss * (1.f / Dc) + EPSc);

  float xn[4];
#pragma unroll
  for (int i = 0; i < 4; ++i) xn[i] = v[i] * rinv * w[lane + 32 * i];

  const float* cp = cosb + ((size_t)b * Qc + q) * Dc;
  const float* sp = sinb + ((size_t)b * Qc + q) * Dc;
  size_t obase = (((size_t)b * heads + h) * outS + outBase + q) * Dc;
#pragma unroll
  for (int i = 0; i < 4; ++i) {
    int d = lane + 32 * i;
    float rot = (i < 2) ? -xn[i + 2] : xn[i - 2]; // rotate_half: [-x2, x1]
    float o = (xn[i] * cp[d] + rot * sp[d]) * outScale;
    out[obase + d] = (_Float16)o;
  }
}

// ---------------------------------------------------------------------------
// Kernel 3a: f32 caches -> f16 buffers.  K keeps (B,KV,S,D); V goes D-MAJOR
// (B,KV,D,S) so PV B-fragments are k-contiguous.
// ---------------------------------------------------------------------------
__global__ void __launch_bounds__(256)
cvt_cache_kernel(const float* __restrict__ kc, const float* __restrict__ vc,
                 _Float16* __restrict__ Kf, _Float16* __restrict__ Vt) {
  size_t i = (size_t)blockIdx.x * 256 + threadIdx.x;
  size_t total = (size_t)Bc * KVc * Pc * Dc;
  if (i >= total) return;
  int d = i % Dc;
  size_t t = i / Dc;
  int p = t % Pc;
  size_t bh = t / Pc;
  Kf[(bh * Sc + p) * Dc + d] = (_Float16)kc[i];
  Vt[(bh * Dc + d) * Sc + p] = (_Float16)vc[i];
}

// Kernel 3b: new V (B,Q,KV,D) -> Vt[(b,kv,d, P+q)].
__global__ void __launch_bounds__(256)
cvt_vnew_kernel(const float* __restrict__ vraw, _Float16* __restrict__ Vt) {
  size_t i = (size_t)blockIdx.x * 256 + threadIdx.x;
  size_t total = (size_t)Bc * Qc * KVc * Dc;
  if (i >= total) return;
  int d = i % Dc;
  size_t t = i / Dc;
  int h = t % KVc;
  size_t bq = t / KVc;
  int q = bq % Qc;
  int b = bq / Qc;
  Vt[(((size_t)b * KVc + h) * Dc + d) * Sc + Pc + q] = (_Float16)vraw[i];
}

// ---------------------------------------------------------------------------
// Kernel 4: flash attention.  2 waves/block; each wave owns a 16-row query
// tile.  Per 32-key block the TDM DMAs the K tile (32x128) and the d-major V
// tile (128x32) into LDS (TENSORcnt), then 8 score WMMAs, causal-masked online
// softmax, P->LDS->A-frag (s_wait_dscnt), 8 PV WMMAs.
// ---------------------------------------------------------------------------
__global__ void __launch_bounds__(64)
attn_fa_kernel(const _Float16* __restrict__ qh, const _Float16* __restrict__ Kf,
               const _Float16* __restrict__ Vt, float* __restrict__ attn) {
  constexpr int QT = Qc / 16; // 64 query tiles per head
  __shared__ _Float16 Kl[2][32][Dc];  // 16 KB: [key][d]
  __shared__ _Float16 Vl[2][Dc][32];  // 16 KB: [d][key]
  __shared__ _Float16 Pl[2][16][32];  //  2 KB

  int wv = threadIdx.x >> 5;
  int lane = threadIdx.x & 31;
  int w = blockIdx.x * 2 + wv;
  int qt = w % QT;
  int h  = (w / QT) % Hc;
  int b  = w / (QT * Hc);
  int kvh = h / Gc;
  int qBase = qt * 16;

  const _Float16* qptr  = qh + (((size_t)b * Hc + h) * Qc + qBase) * Dc;
  const _Float16* kbase = Kf + ((size_t)b * KVc + kvh) * (size_t)Sc * Dc;
  const _Float16* vbase = Vt + ((size_t)b * KVc + kvh) * (size_t)Dc * Sc;
  unsigned kl_off = lds_offset_of(&Kl[wv][0][0]);
  unsigned vl_off = lds_offset_of(&Vl[wv][0][0]);

  v16h aQ[4];
#pragma unroll
  for (int c = 0; c < 4; ++c) aQ[c] = load_a_frag16(qptr + c * 32, Dc);

  v8f accO[8] = {};
  float rowm[8], rowl[8];
#pragma unroll
  for (int r = 0; r < 8; ++r) { rowm[r] = -1e30f; rowl[r] = 0.f; }

  int nn = lane & 15;
  int mh = (lane >> 4) * 8;
  int sEnd = Pc + qBase + 16; // causal horizon (exclusive)

  for (int s0 = 0; s0 < sEnd; s0 += 32) {
    // make sure last iteration's LDS reads retired before the DMA overwrites
    asm volatile("s_wait_dscnt 0" ::: "memory");
    // K tile: 128 contiguous d x 32 key-rows, row stride Dc
    tdm_load_2d_f16(kl_off,
                    (unsigned long long)(uintptr_t)(kbase + (size_t)s0 * Dc),
                    Dc, Sc, Dc, 32, Dc);
    // V tile (d-major): 32 contiguous s x 128 d-rows, row stride Sc
    tdm_load_2d_f16(vl_off,
                    (unsigned long long)(uintptr_t)(vbase + s0),
                    Sc, Dc, 32, Dc, Sc);
    __builtin_amdgcn_s_wait_tensorcnt(0);

    // ---- scores: two 16x16 tiles over keys [s0, s0+32) ----
    v8f sc0 = {}, sc1 = {};
#pragma unroll
    for (int c = 0; c < 4; ++c) {
      v16h b0 = load_bT_frag(&Kl[wv][0][c * 32], Dc);
      v16h b1 = load_bT_frag(&Kl[wv][16][c * 32], Dc);
      sc0 = wmma_f16(aQ[c], b0, sc0);
      sc1 = wmma_f16(aQ[c], b1, sc1);
    }
    // ---- causal mask: key s allowed iff s <= P + qrow ----
#pragma unroll
    for (int r = 0; r < 8; ++r) {
      int lim = Pc + qBase + mh + r;
      if (s0 + nn > lim)      sc0[r] = -1e30f;
      if (s0 + 16 + nn > lim) sc1[r] = -1e30f;
    }
    // ---- online softmax (reduce across the 16-lane C-fragment halves) ----
    float pm[8];
#pragma unroll
    for (int r = 0; r < 8; ++r) pm[r] = fmaxf(sc0[r], sc1[r]);
#pragma unroll
    for (int off = 1; off <= 8; off <<= 1)
#pragma unroll
      for (int r = 0; r < 8; ++r) pm[r] = fmaxf(pm[r], __shfl_xor(pm[r], off, 32));

    float psum[8], corr[8];
#pragma unroll
    for (int r = 0; r < 8; ++r) {
      float mo = rowm[r];
      float mn = fmaxf(mo, pm[r]);
      rowm[r] = mn;
      corr[r] = __expf(mo - mn);
      sc0[r] = __expf(sc0[r] - mn);
      sc1[r] = __expf(sc1[r] - mn);
      psum[r] = sc0[r] + sc1[r];
    }
#pragma unroll
    for (int off = 1; off <= 8; off <<= 1)
#pragma unroll
      for (int r = 0; r < 8; ++r) psum[r] += __shfl_xor(psum[r], off, 32);
#pragma unroll
    for (int r = 0; r < 8; ++r) {
      rowl[r] = rowl[r] * corr[r] + psum[r];
#pragma unroll
      for (int nt = 0; nt < 8; ++nt) accO[nt][r] *= corr[r];
    }
    // ---- P (C-layout) -> LDS -> A-fragment (per-wave patch) ----
#pragma unroll
    for (int r = 0; r < 8; ++r) {
      Pl[wv][mh + r][nn]      = (_Float16)sc0[r];
      Pl[wv][mh + r][16 + nn] = (_Float16)sc1[r];
    }
    asm volatile("s_wait_dscnt 0" ::: "memory"); // wave-local LDS fence
    v16h aP = load_a_frag16(&Pl[wv][0][0], 32);
    // ---- PV: out(16x128) += P(16x32) * V(32x128), V tile is d-major ----
#pragma unroll
    for (int nt = 0; nt < 8; ++nt) {
      v16h bv = load_bT_frag(&Vl[wv][nt * 16][0], 32);
      accO[nt] = wmma_f16(aP, bv, accO[nt]);
    }
  }

  // ---- epilogue: out / rowsum, write attn[b][q][h][d] (f32) ----
#pragma unroll
  for (int nt = 0; nt < 8; ++nt)
#pragma unroll
    for (int r = 0; r < 8; ++r) {
      int qrow = qBase + mh + r;
      attn[(((size_t)b * Qc + qrow) * Hc + h) * Dc + nt * 16 + nn] =
          accO[nt][r] / rowl[r];
    }
}

// ---------------------------------------------------------------------------
// Host-side launch
// ---------------------------------------------------------------------------
extern "C" void kernel_launch(void* const* d_in, const int* in_sizes, int n_in,
                              void* d_out, int out_size, void* d_ws, size_t ws_size,
                              hipStream_t stream) {
  (void)in_sizes; (void)n_in; (void)out_size; (void)ws_size;
  const float* hs      = (const float*)d_in[0];
  const float* cosb    = (const float*)d_in[1];
  const float* sinb    = (const float*)d_in[2];
  const float* k_cache = (const float*)d_in[3];
  const float* v_cache = (const float*)d_in[4];
  const float* Wq      = (const float*)d_in[5];
  const float* Wk      = (const float*)d_in[6];
  const float* Wv      = (const float*)d_in[7];
  const float* Wo      = (const float*)d_in[8];
  const float* qnw     = (const float*)d_in[9];
  const float* knw     = (const float*)d_in[10];

  const int M = Bc * Qc; // 2048

  // Workspace layout (bytes)
  char* ws = (char*)d_ws;
  float*    q_raw = (float*)(ws + 0);                       // 32 MB  (reused as attn)
  float*    k_raw = (float*)(ws + (size_t)33554432);        //  8 MB
  float*    v_raw = (float*)(ws + (size_t)41943040);        //  8 MB
  _Float16* qh    = (_Float16*)(ws + (size_t)50331648);     // 16 MB  (B,H,Q,D)
  _Float16* Kf    = (_Float16*)(ws + (size_t)67108864);     // 16 MB  (B,KV,S,D)
  _Float16* Vt    = (_Float16*)(ws + (size_t)83886080);     // 16 MB  (B,KV,D,S)
  float*    attn  = q_raw;                                  // alias: q_raw dead by then

  dim3 blk(256);
  // 1) QKV projections
  gemm_f16_wmma<<<dim3(Hc * Dc / 128, M / 64), blk, 0, stream>>>(hs, Wq, q_raw, M, HIDc, Hc * Dc);
  gemm_f16_wmma<<<dim3(KVc * Dc / 128, M / 64), blk, 0, stream>>>(hs, Wk, k_raw, M, HIDc, KVc * Dc);
  gemm_f16_wmma<<<dim3(KVc * Dc / 128, M / 64), blk, 0, stream>>>(hs, Wv, v_raw, M, HIDc, KVc * Dc);

  // 2) caches -> f16 K (s-major) / V (d-major)
  {
    size_t total = (size_t)Bc * KVc * Pc * Dc;
    cvt_cache_kernel<<<dim3((total + 255) / 256), blk, 0, stream>>>(k_cache, v_cache, Kf, Vt);
  }
  // 3) RMSNorm + RoPE (q pre-scaled by 1/sqrt(D)); new K into Kf[P..S)
  {
    int nrq = Bc * Qc * Hc;
    norm_rope_kernel<<<dim3(nrq / 8), blk, 0, stream>>>(q_raw, cosb, sinb, qnw, qh,
                                                        Hc, Qc, 0, SCALEc, nrq);
    int nrk = Bc * Qc * KVc;
    norm_rope_kernel<<<dim3(nrk / 8), blk, 0, stream>>>(k_raw, cosb, sinb, knw, Kf,
                                                        KVc, Sc, Pc, 1.0f, nrk);
  }
  // 4) new V into Vt[.., P+q]
  {
    size_t total = (size_t)Bc * Qc * KVc * Dc;
    cvt_vnew_kernel<<<dim3((total + 255) / 256), blk, 0, stream>>>(v_raw, Vt);
  }
  // 5) flash attention (4096 waves, 2 per block, TDM-fed)
  attn_fa_kernel<<<dim3(Bc * Hc * (Qc / 16) / 2), dim3(64), 0, stream>>>(qh, Kf, Vt, attn);

  // 6) output projection -> d_out
  gemm_f16_wmma<<<dim3(HIDc / 128, M / 64), blk, 0, stream>>>(attn, Wo, (float*)d_out,
                                                              M, Hc * Dc, HIDc);
}
